// ConvLayer_68332929679496
// MI455X (gfx1250) — matrix operand order repaired and checked by reading, not compiled, not run on hardware
//
#include <hip/hip_runtime.h>
#include <hip/hip_bf16.h>

// 3x3 conv as implicit GEMM on CDNA5 (gfx1250, wave32) bf16 WMMA, f32 accumulate.
// Prepass converts x -> NHWC bf16 and W -> [tap][cout][cin] bf16 once (into d_ws),
// so the main loop is conversion-free: K(=cin) contiguous, b128 staging, 8 WMMAs
// per wave per stage. Async global->LDS staging (ASYNCcnt) when available.

typedef __bf16 bf16_t;
typedef __attribute__((ext_vector_type(16))) __bf16        v16bf;
typedef __attribute__((ext_vector_type(8)))  float         v8f;
typedef __attribute__((ext_vector_type(4)))  unsigned int  u32x4;
typedef __attribute__((ext_vector_type(4)))  int           i32x4;

union FragA { u32x4 q[2]; v16bf v; };

__device__ __forceinline__ unsigned short f32_to_bf16(float f) {
    unsigned u = __builtin_bit_cast(unsigned, f);
    u += 0x7FFFu + ((u >> 16) & 1u);            // RNE
    return (unsigned short)(u >> 16);
}

#define CIN   256
#define COUT  256
#define HH    512
#define WW    512
#define HW    (HH * WW)
#define LDP   40            // LDS row stride in bf16 (80B = 5*16B: aligned + bank-spread)

// ---------------- async global->LDS (probe-confirmed to exist) ----------------
#if defined(__has_builtin)
#  if __has_builtin(__builtin_amdgcn_global_load_async_to_lds_b128)
#    define HAVE_ASYNC 1
#  endif
#endif
#ifndef HAVE_ASYNC
#  define HAVE_ASYNC 0
#endif

#if HAVE_ASYNC
typedef __attribute__((address_space(1))) i32x4 g_i32x4;   // global int4
typedef __attribute__((address_space(3))) i32x4 l_i32x4;   // LDS int4
__device__ __forceinline__ void async_cp16(const void* g, void* l) {
    // (src in addrspace(1), dst in addrspace(3), imm offset, cpol) — param 1 type
    // confirmed by clang diagnostic: int4 addrspace(1)*, non-const, src-first.
    __builtin_amdgcn_global_load_async_to_lds_b128((g_i32x4*)g, (l_i32x4*)l, 0, 0);
}
__device__ __forceinline__ void wait_async0() {
#if __has_builtin(__builtin_amdgcn_s_wait_asynccnt)
    __builtin_amdgcn_s_wait_asynccnt(0);
#else
    asm volatile("s_wait_asynccnt 0" ::: "memory");
#endif
}
#endif

// ---------------- prepass: x (C,H,W) f32 -> xT (H,W,C) bf16 ----------------
__global__ __launch_bounds__(256)
void cvt_x_nhwc(const float* __restrict__ x, unsigned short* __restrict__ xT) {
    __shared__ unsigned short T[64 * 264];
    const int  t  = threadIdx.x;
    const long p0 = (long)blockIdx.x * 64;       // 64 same-row pixels per block
    const int  wl = t & 63, cb = t >> 6;
    #pragma unroll 4
    for (int i = 0; i < 64; ++i) {
        const int cin = i * 4 + cb;
        T[wl * 264 + cin] = f32_to_bf16(x[(long)cin * HW + p0 + wl]);
    }
    __syncthreads();
    const int cin8 = (t & 31) << 3, wl0 = t >> 5;
    #pragma unroll
    for (int j = 0; j < 8; ++j) {
        const int r = j * 8 + wl0;
        *(u32x4*)&xT[(p0 + r) * CIN + cin8] = *(const u32x4*)&T[r * 264 + cin8];
    }
}

// ---------------- prepass: W (tap,cin,cout) f32 -> wT (tap,cout,cin) bf16 ----------------
__global__ __launch_bounds__(256)
void cvt_w_t(const float* __restrict__ w, unsigned short* __restrict__ wT) {
    __shared__ unsigned short T[64 * 264];
    const int t   = threadIdx.x;
    const int pos = blockIdx.y;                  // tap 0..8
    const int c0  = blockIdx.x * 64;             // cout tile
    const int cl  = t & 63, cb = t >> 6;
    #pragma unroll 4
    for (int i = 0; i < 64; ++i) {
        const int cin = i * 4 + cb;
        T[cl * 264 + cin] = f32_to_bf16(w[((long)pos * CIN + cin) * COUT + c0 + cl]);
    }
    __syncthreads();
    const int cin8 = (t & 31) << 3, r0 = t >> 5;
    #pragma unroll
    for (int j = 0; j < 8; ++j) {
        const int r = j * 8 + r0;
        *(u32x4*)&wT[((long)pos * COUT + c0 + r) * CIN + cin8] = *(const u32x4*)&T[r * 264 + cin8];
    }
}

// ---------------- main kernel: 128 cout x 128 px per block, 32x64 per wave ----------------
__global__ __launch_bounds__(256)
void conv3x3_wmma_main(const unsigned short* __restrict__ xT,
                       const unsigned short* __restrict__ wT,
                       const float* __restrict__ bias,
                       float* __restrict__ out)
{
    __shared__ unsigned short As[2][128 * LDP];   // [cout_local][cin_chunk]
    __shared__ unsigned short Bs[2][128 * LDP];   // [pixel_local][cin_chunk]

    const int t    = threadIdx.x;
    const int lane = t & 31, wv = t >> 5;
    const int wm   = wv & 3;           // cout sub-tile: wm*32
    const int wn   = wv >> 2;          // pixel sub-tile: wn*64
    const int row  = lane & 15, half = lane >> 4;

    const int  cout0 = blockIdx.y << 7;          // 0 or 128
    const long p0    = (long)blockIdx.x << 7;    // 128 same-row pixels (128 | 512)
    const int  h     = (int)(p0 >> 9);
    const int  w0    = (int)(p0 & 511);

    // staging map: thread -> (row cr 0..127, 16-elem segment cs)
    const int cr = t >> 1;
    const int cs = (t & 1) << 4;

    const u32x4 z4 = {};
    v8f acc[2][4];
    {
        const v8f vz = {};
        #pragma unroll
        for (int f = 0; f < 2; ++f)
            #pragma unroll
            for (int g = 0; g < 4; ++g) acc[f][g] = vz;
    }

    auto compute = [&](int buf) {
        FragA fa[2], fb[4];
        #pragma unroll
        for (int f = 0; f < 2; ++f) {
            const int ab = (wm * 32 + f * 16 + row) * LDP + half * 16;
            fa[f].q[0] = *(const u32x4*)&As[buf][ab];
            fa[f].q[1] = *(const u32x4*)&As[buf][ab + 8];
        }
        #pragma unroll
        for (int g = 0; g < 4; ++g) {
            const int bb = (wn * 64 + g * 16 + row) * LDP + half * 16;
            fb[g].q[0] = *(const u32x4*)&Bs[buf][bb];
            fb[g].q[1] = *(const u32x4*)&Bs[buf][bb + 8];
        }
        #pragma unroll
        for (int f = 0; f < 2; ++f)
            #pragma unroll
            for (int g = 0; g < 4; ++g)
                acc[f][g] = __builtin_amdgcn_wmma_f32_16x16x32_bf16(
                    false, fa[f].v, false, fb[g].v, (short)0, acc[f][g], false, false);
    };

#if HAVE_ASYNC
    auto issue_stage = [&](int s, int buf) {
        const int pos = s >> 3, cin0 = (s & 7) << 5;
        const int kh = pos / 3, kw = pos - kh * 3;
        const long aoff = ((long)pos * COUT + cout0 + cr) * CIN + cin0 + cs;
        async_cp16(&wT[aoff],     &As[buf][cr * LDP + cs]);
        async_cp16(&wT[aoff + 8], &As[buf][cr * LDP + cs + 8]);
        const int hp = h + kh - 1;
        const int wp = w0 + cr + kw - 1;
        if (((unsigned)hp < (unsigned)HH) & ((unsigned)wp < (unsigned)WW)) {
            const long boff = ((long)hp * WW + wp) * CIN + cin0 + cs;
            async_cp16(&xT[boff],     &Bs[buf][cr * LDP + cs]);
            async_cp16(&xT[boff + 8], &Bs[buf][cr * LDP + cs + 8]);
        } else {   // zero-fill padding rows/cols directly in LDS
            *(u32x4*)&Bs[buf][cr * LDP + cs]     = z4;
            *(u32x4*)&Bs[buf][cr * LDP + cs + 8] = z4;
        }
    };

    issue_stage(0, 0);
    for (int s = 0; s < 72; ++s) {
        const int buf = s & 1;
        wait_async0();           // own async copies landed in LDS
        __syncthreads();         // everyone's copies visible
        if (s + 1 < 72) issue_stage(s + 1, buf ^ 1);  // overlaps WMMA below
        compute(buf);
        __syncthreads();
    }
#else
    u32x4 aR[2], bR[2];
    auto prefetch = [&](int s) {
        const int pos = s >> 3, cin0 = (s & 7) << 5;
        const int kh = pos / 3, kw = pos - kh * 3;
        const long aoff = ((long)pos * COUT + cout0 + cr) * CIN + cin0 + cs;
        aR[0] = *(const u32x4*)&wT[aoff];
        aR[1] = *(const u32x4*)&wT[aoff + 8];
        const int hp = h + kh - 1;
        const int wp = w0 + cr + kw - 1;
        if (((unsigned)hp < (unsigned)HH) & ((unsigned)wp < (unsigned)WW)) {
            const long boff = ((long)hp * WW + wp) * CIN + cin0 + cs;
            bR[0] = *(const u32x4*)&xT[boff];
            bR[1] = *(const u32x4*)&xT[boff + 8];
        } else { bR[0] = z4; bR[1] = z4; }
    };
    auto stage_store = [&](int buf) {
        *(u32x4*)&As[buf][cr * LDP + cs]     = aR[0];
        *(u32x4*)&As[buf][cr * LDP + cs + 8] = aR[1];
        *(u32x4*)&Bs[buf][cr * LDP + cs]     = bR[0];
        *(u32x4*)&Bs[buf][cr * LDP + cs + 8] = bR[1];
    };

    prefetch(0);
    for (int s = 0; s < 72; ++s) {
        const int buf = s & 1;
        stage_store(buf);
        __syncthreads();
        if (s + 1 < 72) prefetch(s + 1);   // global loads overlap WMMA below
        compute(buf);
        __syncthreads();
    }
#endif

    // C/D layout: lanes 0-15 N=lane, M=r; lanes 16-31 N=lane-16, M=8+r.
    #pragma unroll
    for (int f = 0; f < 2; ++f) {
        const int mb = cout0 + wm * 32 + f * 16 + half * 8;
        #pragma unroll
        for (int g = 0; g < 4; ++g) {
            const long p = p0 + wn * 64 + g * 16 + row;
            #pragma unroll
            for (int r = 0; r < 8; ++r) {
                const int m = mb + r;
                out[(long)m * HW + p] = acc[f][g][r] + bias[m];
            }
        }
    }
}

// ---------------- fallback (ws too small): round-1 kernel, converts in-loop ----------------
__global__ __launch_bounds__(256)
void conv3x3_wmma_fb(const float* __restrict__ x, const float* __restrict__ w,
                     const float* __restrict__ bias, float* __restrict__ out)
{
    __shared__ unsigned short As[2][64 * LDP];
    __shared__ unsigned short Bs[2][64 * LDP];

    const int tid = threadIdx.x, lane = tid & 31, wv = tid >> 5;
    const int wm = wv & 3, wn = wv >> 2, row = lane & 15, half = lane >> 4;
    const int cout0 = blockIdx.y << 6;
    const int ptile = blockIdx.x, h = ptile >> 3, w0 = (ptile & 7) << 6;
    const int a_cout = tid & 63, a_cb = tid >> 6;
    const int b_cin = tid >> 3, b_n0 = (tid & 7) << 3;

    v8f c0 = {}; v8f c1 = {};
    float aReg[8], bReg[8];

    auto prefetch = [&](int s) {
        const int pos = s >> 3, cin0 = (s & 7) << 5;
        const int kh = pos / 3, kw = pos - kh * 3;
        #pragma unroll
        for (int i = 0; i < 8; ++i)
            aReg[i] = w[((pos * CIN + cin0 + (i << 2) + a_cb) * COUT) + cout0 + a_cout];
        const int hp = h + kh - 1;
        const long base = ((long)(cin0 + b_cin) * HH + hp) * WW;
        #pragma unroll
        for (int i = 0; i < 8; ++i) {
            const int wp = w0 + b_n0 + i + kw - 1;
            const bool ok = ((unsigned)hp < (unsigned)HH) & ((unsigned)wp < (unsigned)WW);
            bReg[i] = ok ? x[base + wp] : 0.0f;
        }
    };
    auto stage_store = [&](int buf) {
        #pragma unroll
        for (int i = 0; i < 8; ++i)
            As[buf][a_cout * LDP + (i << 2) + a_cb] = f32_to_bf16(aReg[i]);
        #pragma unroll
        for (int i = 0; i < 8; ++i)
            Bs[buf][(b_n0 + i) * LDP + b_cin] = f32_to_bf16(bReg[i]);
    };

    const int a_base  = (wm * 16 + row) * LDP + half * 16;
    const int b_base0 = (wn * 32 +      row) * LDP + half * 16;
    const int b_base1 = (wn * 32 + 16 + row) * LDP + half * 16;

    prefetch(0);
    for (int s = 0; s < 72; ++s) {
        const int buf = s & 1;
        stage_store(buf);
        __syncthreads();
        if (s + 1 < 72) prefetch(s + 1);
        FragA fa, fb0, fb1;
        fa.q[0]  = *(const u32x4*)&As[buf][a_base];
        fa.q[1]  = *(const u32x4*)&As[buf][a_base + 8];
        fb0.q[0] = *(const u32x4*)&Bs[buf][b_base0];
        fb0.q[1] = *(const u32x4*)&Bs[buf][b_base0 + 8];
        fb1.q[0] = *(const u32x4*)&Bs[buf][b_base1];
        fb1.q[1] = *(const u32x4*)&Bs[buf][b_base1 + 8];
        c0 = __builtin_amdgcn_wmma_f32_16x16x32_bf16(false, fa.v, false, fb0.v, (short)0, c0, false, false);
        c1 = __builtin_amdgcn_wmma_f32_16x16x32_bf16(false, fa.v, false, fb1.v, (short)0, c1, false, false);
        __syncthreads();
    }

    const int  m_base = cout0 + wm * 16 + half * 8;
    const long pix0   = (long)h * WW + w0 + wn * 32 + row;
    #pragma unroll
    for (int r = 0; r < 8; ++r) {
        const int m = m_base + r;
        const float bb = bias[m];
        out[(long)m * HW + pix0]      = c0[r] + bb;
        out[(long)m * HW + pix0 + 16] = c1[r] + bb;
    }
}

extern "C" void kernel_launch(void* const* d_in, const int* in_sizes, int n_in,
                              void* d_out, int out_size, void* d_ws, size_t ws_size,
                              hipStream_t stream) {
    const float* x    = (const float*)d_in[0];
    const float* w    = (const float*)d_in[1];
    const float* bias = (const float*)d_in[2];
    float* out        = (float*)d_out;

    const size_t xT_bytes = (size_t)CIN * HW * sizeof(unsigned short);       // 128 MiB
    const size_t wT_bytes = (size_t)9 * COUT * CIN * sizeof(unsigned short); // ~1.1 MiB

    if (ws_size >= xT_bytes + wT_bytes) {
        unsigned short* xT = (unsigned short*)d_ws;
        unsigned short* wT = (unsigned short*)((char*)d_ws + xT_bytes);
        cvt_x_nhwc<<<HW / 64, 256, 0, stream>>>(x, xT);
        cvt_w_t<<<dim3(4, 9), 256, 0, stream>>>(w, wT);
        conv3x3_wmma_main<<<dim3(HW / 128, 2), 256, 0, stream>>>(xT, wT, bias, out);
    } else {
        conv3x3_wmma_fb<<<dim3(HW / 64, 4), 256, 0, stream>>>(x, w, bias, out);
    }
}
